// DiagonalSelectiveCell_2396591751752
// MI455X (gfx1250) — matrix operand order repaired and checked by reading, not compiled, not run on hardware
//
#include <hip/hip_runtime.h>
#include <cstddef>
#include <cstdint>

typedef __bf16 bf16;
typedef __attribute__((ext_vector_type(16))) __bf16 v16bf;
typedef __attribute__((ext_vector_type(8)))  __bf16 v8bf;
typedef __attribute__((ext_vector_type(8)))  float  v8f;
typedef __attribute__((ext_vector_type(4)))  int    v4i;

#if __has_builtin(__builtin_amdgcn_global_load_async_to_lds_b128) && \
    __has_builtin(__builtin_amdgcn_s_wait_asynccnt)
#define HAS_ASYNC 1
#else
#define HAS_ASYNC 0
#endif

typedef __attribute__((address_space(1))) v4i* gptr_b128;   // global int4*
typedef __attribute__((address_space(3))) v4i* lptr_b128;   // LDS int4*

// Problem sizes (fixed by reference)
constexpr int T_ = 2048;
constexpr int B_ = 16;
constexpr int D_ = 1024;
constexpr int M_ = T_ * B_;   // 32768 rows
constexpr int K_ = D_;
constexpr int N_ = D_;

// GEMM tiling
constexpr int BM = 128;
constexpr int BN = 64;
constexpr int BK = 32;
constexpr int NK = K_ / BK;
constexpr int LDA = BK + 8;   // 40 bf16 = 80B row stride -> 16B aligned, conflict-free b128
constexpr int LDB = BK + 8;

// ---------------------------------------------------------------------------
// Split f32 -> bf16 hi + bf16 lo  (x ~= hi + lo, ~16 mantissa bits total)
// ---------------------------------------------------------------------------
__global__ __launch_bounds__(256) void convert_hilo(const float* __restrict__ src,
                                                    bf16* __restrict__ hi,
                                                    bf16* __restrict__ lo,
                                                    int n) {
    int i = blockIdx.x * 256 + threadIdx.x;
    if (i < n) {
        float v = src[i];
        bf16 h = (bf16)v;
        hi[i] = h;
        lo[i] = (bf16)(v - (float)h);
    }
}

// 16B global -> LDS copy: async (no VGPR staging, ASYNCcnt) if available.
__device__ __forceinline__ void copy_b128(const bf16* g, bf16* l) {
#if HAS_ASYNC
    __builtin_amdgcn_global_load_async_to_lds_b128(
        (gptr_b128)(unsigned long long)(const void*)g,
        (lptr_b128)(unsigned int)(unsigned long long)(void*)l,
        0, 0);
#else
    *(uint4*)l = *(const uint4*)g;
#endif
}

__device__ __forceinline__ void async_wait_all() {
#if HAS_ASYNC
    __builtin_amdgcn_s_wait_asynccnt(0);
#endif
}

// Build a 16x32 (A) or 32x16 (B) bf16 fragment from two 16B LDS reads.
__device__ __forceinline__ v16bf frag16(const bf16* p0, const bf16* p1) {
    v8bf a = *(const v8bf*)p0;
    v8bf b = *(const v8bf*)p1;
    v16bf r;
#pragma unroll
    for (int i = 0; i < 8; ++i) { r[i] = a[i]; r[i + 8] = b[i]; }
    return r;
}

// ---------------------------------------------------------------------------
// Fused triple GEMM:  out[mat][m][e] = act( sum_d x[m][d] * W[mat][e][d] + bias[mat][e] )
//   mat 0: delta  = sigmoid(.)  -> ws
//   mat 1: cand_x = (.)         -> d_out h-region (h[1:])
//   mat 2: gate   = silu(.)     -> d_out output region
// Split-precision: acc += a_lo*b_hi + a_hi*b_lo + a_hi*b_hi  (3x bf16 WMMA)
// Double-buffered LDS; async global->LDS copies overlap next tile with WMMAs.
// ---------------------------------------------------------------------------
__global__ __launch_bounds__(256) void gemm3(const bf16* __restrict__ xhi,
                                             const bf16* __restrict__ xlo,
                                             const bf16* __restrict__ whi,   // [3][D][D]
                                             const bf16* __restrict__ wlo,   // [3][D][D]
                                             const float* __restrict__ bias0,
                                             const float* __restrict__ bias1,
                                             const float* __restrict__ bias2,
                                             float* __restrict__ out0,
                                             float* __restrict__ out1,
                                             float* __restrict__ out2) {
    __shared__ alignas(16) bf16 As[2][2][BM][LDA];      // [buf][hi/lo][row][k]
    __shared__ alignas(16) bf16 Bs[2][3][2][BN][LDB];   // [buf][mat][hi/lo][e][k]

    const int tid  = threadIdx.x;
    const int wave = tid >> 5;
    const int lane = tid & 31;
    const int wm   = wave >> 1;            // 0..3 -> 32-row strip
    const int wn   = wave & 1;             // 0..1 -> 32-col strip
    const int bm   = blockIdx.x * BM;
    const int bn   = blockIdx.y * BN;
    const int l15  = lane & 15;
    const bool hiHalf = (lane >= 16);

    v8f acc[3][2][2] = {};

    // Issue all global->LDS copies for one K-tile into LDS buffer `buf`.
    auto fill = [&](int buf, int k0) {
#pragma unroll
        for (int it = 0; it < 2; ++it) {
            int c   = tid + it * 256;          // 0..511
            int row = c >> 2;
            int cc  = c & 3;
            size_t g = (size_t)(bm + row) * K_ + k0 + cc * 8;
            copy_b128(xhi + g, &As[buf][0][row][cc * 8]);
            copy_b128(xlo + g, &As[buf][1][row][cc * 8]);
        }
#pragma unroll
        for (int mp = 0; mp < 6; ++mp) {
            int mat  = mp >> 1;
            int prec = mp & 1;
            int row  = tid >> 2;
            int cc   = tid & 3;
            const bf16* wsrc = (prec ? wlo : whi) + (size_t)mat * D_ * D_ +
                               (size_t)(bn + row) * D_ + k0 + cc * 8;
            copy_b128(wsrc, &Bs[buf][mat][prec][row][cc * 8]);
        }
    };

    fill(0, 0);   // prologue

    for (int i = 0; i < NK; ++i) {
        const int buf = i & 1;
        async_wait_all();     // own copies into buf are complete
        __syncthreads();      // everyone's copies done; everyone done reading buf^1
        if (i + 1 < NK) fill(buf ^ 1, (i + 1) * BK);   // overlaps with WMMAs below

        // ---- A fragments (16x32 bf16, contiguous k order used here (B matches): lanes<16 hold K0..15, lanes>=16 hold K16..31) ----
        v16bf a_hi[2], a_lo[2];
#pragma unroll
        for (int ms = 0; ms < 2; ++ms) {
            int row = wm * 32 + ms * 16 + l15;
            int kb  = hiHalf ? 16 : 0;
            a_hi[ms] = frag16(&As[buf][0][row][kb], &As[buf][0][row][kb + 8]);
            a_lo[ms] = frag16(&As[buf][1][row][kb], &As[buf][1][row][kb + 8]);
        }

#pragma unroll
        for (int mat = 0; mat < 3; ++mat) {
            // B fragments (32x16 bf16: lane n holds col e0+n, K consecutive)
            v16bf bh[2], bl[2];
#pragma unroll
            for (int ns = 0; ns < 2; ++ns) {
                int row = wn * 32 + ns * 16 + l15;
                int kb  = hiHalf ? 16 : 0;
                bh[ns] = frag16(&Bs[buf][mat][0][row][kb], &Bs[buf][mat][0][row][kb + 8]);
                bl[ns] = frag16(&Bs[buf][mat][1][row][kb], &Bs[buf][mat][1][row][kb + 8]);
            }
#pragma unroll
            for (int ms = 0; ms < 2; ++ms) {
#pragma unroll
                for (int ns = 0; ns < 2; ++ns) {
                    v8f c = acc[mat][ms][ns];
                    c = __builtin_amdgcn_wmma_f32_16x16x32_bf16(
                            false, a_lo[ms], false, bh[ns], (short)0, c, false, false);
                    c = __builtin_amdgcn_wmma_f32_16x16x32_bf16(
                            false, a_hi[ms], false, bl[ns], (short)0, c, false, false);
                    c = __builtin_amdgcn_wmma_f32_16x16x32_bf16(
                            false, a_hi[ms], false, bh[ns], (short)0, c, false, false);
                    acc[mat][ms][ns] = c;
                }
            }
        }
    }

    // ---- epilogue: bias + activation + store ----
    const float* biases[3] = { bias0, bias1, bias2 };
    float*       dsts[3]   = { out0,  out1,  out2  };
#pragma unroll
    for (int mat = 0; mat < 3; ++mat) {
#pragma unroll
        for (int ns = 0; ns < 2; ++ns) {
            int e = bn + wn * 32 + ns * 16 + l15;
            float bias = biases[mat][e];
#pragma unroll
            for (int ms = 0; ms < 2; ++ms) {
                int mbase = bm + wm * 32 + ms * 16 + (hiHalf ? 8 : 0);
#pragma unroll
                for (int j = 0; j < 8; ++j) {
                    float v = acc[mat][ms][ns][j] + bias;
                    if (mat == 0)      v = 1.0f / (1.0f + __expf(-v));      // sigmoid
                    else if (mat == 2) v = v / (1.0f + __expf(-v));         // silu
                    dsts[mat][(size_t)(mbase + j) * N_ + e] = v;
                }
            }
        }
    }
}

// ---------------------------------------------------------------------------
// Sequential scan over T; one thread per (b,d).
// ---------------------------------------------------------------------------
__global__ __launch_bounds__(256) void scan_kernel(const float* __restrict__ delta,
                                                   const float* __restrict__ h0,
                                                   const float* __restrict__ r_h,
                                                   float* __restrict__ out_region,
                                                   float* __restrict__ h_region) {
    int i = blockIdx.x * 256 + threadIdx.x;        // 0 .. B*D-1
    float h = h0[i];
    float r = r_h[i & (D_ - 1)];
    h_region[i] = h;                                // h[0]
    float* hseq = h_region + B_ * D_;               // h[1:]
    for (int t = 0; t < T_; ++t) {
        size_t off = (size_t)t * (B_ * D_) + i;
        float d  = delta[off];
        float cx = hseq[off];
        float g  = out_region[off];
        float cand = tanhf(fmaf(r, h, cx));
        h = fmaf(d, cand - h, h);                   // (1-d)*h + d*cand
        hseq[off]       = h;
        out_region[off] = h * g;
    }
}

// ---------------------------------------------------------------------------
extern "C" void kernel_launch(void* const* d_in, const int* in_sizes, int n_in,
                              void* d_out, int out_size, void* d_ws, size_t ws_size,
                              hipStream_t stream) {
    const float* x       = (const float*)d_in[0];
    const float* h0      = (const float*)d_in[1];
    const float* W_x     = (const float*)d_in[2];
    const float* r_h     = (const float*)d_in[3];
    const float* W_delta = (const float*)d_in[4];
    const float* W_gate  = (const float*)d_in[5];
    const float* b       = (const float*)d_in[6];
    const float* b_delta = (const float*)d_in[7];
    const float* b_gate  = (const float*)d_in[8];

    float* out_region = (float*)d_out;                       // [T,B,D]
    float* h_region   = out_region + (size_t)T_ * B_ * D_;   // [T+1,B,D]

    // Workspace layout
    char*  ws    = (char*)d_ws;
    float* delta = (float*)ws;                                // T*B*D f32
    bf16*  xhi   = (bf16*)(ws + (size_t)T_ * B_ * D_ * 4);
    bf16*  xlo   = xhi + (size_t)T_ * B_ * D_;
    bf16*  whi   = xlo + (size_t)T_ * B_ * D_;                // [3][D][D]
    bf16*  wlo   = whi + (size_t)3 * D_ * D_;

    const int nx = T_ * B_ * D_;
    const int nw = D_ * D_;
    convert_hilo<<<(nx + 255) / 256, 256, 0, stream>>>(x, xhi, xlo, nx);
    // mat order: 0 = delta, 1 = cand_x, 2 = gate
    convert_hilo<<<(nw + 255) / 256, 256, 0, stream>>>(W_delta, whi,          wlo,          nw);
    convert_hilo<<<(nw + 255) / 256, 256, 0, stream>>>(W_x,     whi + nw,     wlo + nw,     nw);
    convert_hilo<<<(nw + 255) / 256, 256, 0, stream>>>(W_gate,  whi + 2 * nw, wlo + 2 * nw, nw);

    dim3 grid(M_ / BM, N_ / BN);   // 256 x 16
    gemm3<<<grid, 256, 0, stream>>>(xhi, xlo, whi, wlo,
                                    b_delta, b, b_gate,
                                    delta,                     // sigmoid -> ws
                                    h_region + B_ * D_,        // cand_x  -> h[1:]
                                    out_region);               // silu    -> output slot

    scan_kernel<<<(B_ * D_) / 256, 256, 0, stream>>>(delta, h0, r_h, out_region, h_region);
}
